// SIFTDescriptor_71983651881333
// MI455X (gfx1250) — matrix operand order repaired and checked by reading, not compiled, not run on hardware
//
#include <hip/hip_runtime.h>
#include <hip/hip_bf16.h>
#include <math.h>

// ---- CDNA5 vector types ----
typedef __attribute__((ext_vector_type(16))) _Float16 v16h;
typedef __attribute__((ext_vector_type(8)))  _Float16 v8h;
typedef __attribute__((ext_vector_type(8)))  float    v8f;

#define PATCH   32
#define NUM_ANG 8
#define KSIZE   12      // 2*int(32/5)
#define SSTRIDE 8
#define PAD     3
#define CLIPVAL 0.2f
#define EPSV    1e-10f

// LDS layout paddings (halves): 16B-aligned v8h loads + distinct-bank lane strides.
#define BT_STRIDE 1048   // 2096 B/row = 524 dwords; 12*k mod 64 distinct for k=0..15
#define A_STRIDE  40     // 80 B/row   = 20 dwords;  20*k mod 64 distinct for k=0..15

// ---- fast HW math helpers ----
__device__ __forceinline__ float fast_rcp(float x) {
#if __has_builtin(__builtin_amdgcn_rcpf)
    return __builtin_amdgcn_rcpf(x);
#else
    return 1.0f / x;
#endif
}
__device__ __forceinline__ float fast_sqrt(float x) {
#if __has_builtin(__builtin_amdgcn_sqrtf)
    return __builtin_amdgcn_sqrtf(x);
#else
    return sqrtf(x);
#endif
}

// Branchless atan2, ~1e-6 rad (11th-order odd minimax on [0,1] + octant fixup).
__device__ __forceinline__ float fast_atan2(float y, float x) {
    const float ax = fabsf(x), ay = fabsf(y);
    const float mn = fminf(ax, ay);
    const float mx = fmaxf(fmaxf(ax, ay), 1e-30f);
    const float r  = mn * fast_rcp(mx);
    const float s  = r * r;
    float p = -0.01172120f;
    p = fmaf(p, s,  0.05265332f);
    p = fmaf(p, s, -0.11643287f);
    p = fmaf(p, s,  0.19354346f);
    p = fmaf(p, s, -0.33262347f);
    p = fmaf(p, s,  0.99997726f);
    p *= r;
    float a = (ay > ax) ? (1.57079632679489662f - p) : p;
    a = (x < 0.0f) ? (3.14159265358979323f - a) : a;
    return (y < 0.0f) ? -a : a;
}

// xor-permute within a row of 16 lanes + add: one v_permlane16_b32, no LDS.
template <int M>
__device__ __forceinline__ float row_xor_add(float v) {
#if __has_builtin(__builtin_amdgcn_permlane16)
    constexpr unsigned LO = (M == 1) ? 0x67452301u : (M == 2) ? 0x54761032u
                          : (M == 4) ? 0x32107654u : 0xFEDCBA98u;
    constexpr unsigned HI = (M == 1) ? 0xEFCDAB89u : (M == 2) ? 0xDCFE98BAu
                          : (M == 4) ? 0xBA98FEDCu : 0x76543210u;
    int t = __builtin_amdgcn_permlane16(__float_as_int(v), __float_as_int(v),
                                        (int)LO, (int)HI, false, false);
    return v + __int_as_float(t);
#elif __has_builtin(__builtin_amdgcn_ds_swizzle)
    int t = __builtin_amdgcn_ds_swizzle(__float_as_int(v), (M << 10) | 0x1F);
    return v + __int_as_float(t);
#else
    return v + __shfl_xor(v, M, 32);
#endif
}
__device__ __forceinline__ float reduce16(float v) {
    v = row_xor_add<1>(v);
    v = row_xor_add<2>(v);
    v = row_xor_add<4>(v);
    v = row_xor_add<8>(v);
    return v;
}

// 256 threads = 8 waves; each wave processes patch PAIRS (M=16 WMMA rows) in a
// grid-stride loop so the per-block Bt/g1 setup is amortized over ~4 pairs.
__global__ __launch_bounds__(256)
void SIFTDescriptor_71983651881333_kernel(const float* __restrict__ in,
                                          float* __restrict__ out,
                                          int nPatches) {
    __shared__ alignas(16) _Float16 Bt[16][BT_STRIDE];         // W^T: [cell n][pixel k]
    __shared__ alignas(16) _Float16 Ach[8][2][16][A_STRIDE];   // per-wave double-buffered A tile
    __shared__ float g1[PATCH];
    __shared__ float graw[PATCH];

    const int tid    = threadIdx.x;
    const int wave   = tid >> 5;
    const int lane   = tid & 31;
    const int half16 = lane & 15;
    const bool hi    = lane >= 16;

    // ---- one-time per block: normalized 1D gaussian (sigma = 32/sqrt(2)) ----
    if (tid < PATCH) {
        float x = (float)tid - 16.0f + 0.5f;              // even ksize offset
        graw[tid] = __expf(-(x * x) * (1.0f / 1024.0f));  // 2*sigma^2 = 1024
    }
    __syncthreads();
    if (tid < PATCH) {
        float s = 0.0f;
        #pragma unroll
        for (int j = 0; j < PATCH; ++j) s += graw[j];
        g1[tid] = graw[tid] / s;
    }

    // ---- one-time per block: pooling weights W[k=pixel][n=cell] into Bt ----
    for (int idx = tid; idx < 1024 * 16; idx += 256) {
        const int n  = idx & 15;
        const int k  = idx >> 4;
        const int py = k >> 5, px = k & 31;
        const int cy = n >> 2, cx = n & 3;
        const int dy = py - (cy * SSTRIDE - PAD);
        const int dx = px - (cx * SSTRIDE - PAD);
        float w = 0.0f;
        if (dy >= 0 && dy < KSIZE && dx >= 0 && dx < KSIZE) {
            const float ty = 6.0f - fabsf((float)dy + 0.5f - 6.0f);
            const float tx = 6.0f - fabsf((float)dx + 0.5f - 6.0f);
            w = ty * tx * (1.0f / 36.0f);
        }
        Bt[n][k] = (_Float16)w;
    }
    __syncthreads();

    // lane-invariant precomputes
    const float gl = g1[lane];
    const int   xm = (lane == 0)  ? 0  : lane - 1;
    const int   xp = (lane == 31) ? 31 : lane + 1;
    // cooperative A-tile zeroing slot: lane -> (row, 32B half of cols 0..31)
    const int   zrow  = lane >> 1;
    const int   zhalf = (lane & 1) * 16;   // halves offset (16 halves = 32B)

    const int totalPairs = nPatches >> 1;
    const int waveStride = gridDim.x * 8;

    for (int pair = blockIdx.x * 8 + wave; pair < totalPairs; pair += waveStride) {
        const int pbase = pair * 2;
        const float* __restrict__ p0 = in + (size_t)pbase * 1024;
        const float* __restrict__ p1 = p0 + 1024;

        v8f acc = {};                        // 16x16 f32 accumulator

        // ---- K loop: chunk c == patch row c (32 pixels per patch) ----
        #pragma unroll 2
        for (int c = 0; c < 32; ++c) {
            const float gw = g1[c] * gl;
            const int   ym = (c == 0)  ? 0  : c - 1;
            const int   yp = (c == 31) ? 31 : c + 1;
            const int   pf = (c < 30) ? c + 2 : 31;     // branchless near prefetch
            __builtin_prefetch(p0 + pf * 32 + lane, 0, 3);
            __builtin_prefetch(p1 + pf * 32 + lane, 0, 3);

            _Float16 (* __restrict__ At)[A_STRIDE] = Ach[wave][c & 1];

            // cooperative zero of the used 16x32 tile region: 2x ds_store_b128/lane
            {
                const v8h z = {};
                _Float16* zp = &At[zrow][zhalf];
                *(v8h*)zp       = z;
                *(v8h*)(zp + 8) = z;
            }

            #pragma unroll
            for (int pp = 0; pp < 2; ++pp) {
                const float* __restrict__ src = pp ? p1 : p0;
                const float l = src[c * 32 + xm];
                const float r = src[c * 32 + xp];
                const float u = src[ym * 32 + lane];
                const float d = src[yp * 32 + lane];
                const float gx = 0.5f * (r - l);
                const float gy = 0.5f * (d - u);
                const float mag = fast_sqrt(fmaf(gx, gx, fmaf(gy, gy, EPSV))) * gw;
                const float a   = fast_atan2(gy, gx + EPSV);
                const float ob  = fmaf(a, 1.27323954473516268615f, 8.0f); // (a+2pi)*8/(2pi)
                const float fb  = floorf(ob);
                const float w1  = ob - fb;
                const int b0 = ((int)fb) & 7;
                const int b1 = (b0 + 1) & 7;
                // scatter only the two nonzero soft-bin contributions
                At[pp * 8 + b0][lane] = (_Float16)((1.0f - w1) * mag);
                At[pp * 8 + b1][lane] = (_Float16)(w1 * mag);
            }

            // cross-lane RAW through LDS within this wave (zero -> scatter -> load)
            asm volatile("s_wait_dscnt 0" ::: "memory");

            // A fragment: 16x32 f16 (lanes<16: K 0..7 & 16..23; lanes>=16: K 8..15 & 24..31)
            const int k0 = hi ? 8 : 0;
            const _Float16* arow = &At[half16][0];
            const v8h alo = *(const v8h*)(arow + k0);
            const v8h ahi = *(const v8h*)(arow + k0 + 16);
            const v16h a = __builtin_shufflevector(alo, ahi,
                0,1,2,3,4,5,6,7,8,9,10,11,12,13,14,15);

            // B fragment: 32x16 f16 (lanes<16: K 0..7 & 16..23; lanes>=16: K 8..15 & 24..31), col n = lane%16
            const _Float16* brow = &Bt[half16][c * 32 + (hi ? 8 : 0)];
            const v8h blo = *(const v8h*)(brow);
            const v8h bhi = *(const v8h*)(brow + 16);
            const v16h b = __builtin_shufflevector(blo, bhi,
                0,1,2,3,4,5,6,7,8,9,10,11,12,13,14,15);

            acc = __builtin_amdgcn_wmma_f32_16x16x32_f16(
                /*neg_a=*/false, a, /*neg_b=*/false, b,
                /*c_mod=*/(short)0, acc, /*reuse_a=*/false, /*reuse_b=*/false);
        }

        // ---- normalization chain: lanes 0-15 hold patch0, lanes 16-31 patch1 ----
        float ss = 0.0f;
        #pragma unroll
        for (int i = 0; i < 8; ++i) ss = fmaf(acc[i], acc[i], ss);
        ss = reduce16(ss);
        const float inv1 = fast_rcp(fmaxf(fast_sqrt(ss), 1e-12f));

        float d1[8];
        float ss2 = 0.0f;
        #pragma unroll
        for (int i = 0; i < 8; ++i) {
            float t = fminf(fmaxf(acc[i] * inv1, 0.0f), CLIPVAL);
            d1[i] = t;
            ss2 = fmaf(t, t, ss2);
        }
        ss2 = reduce16(ss2);
        const float inv2 = fast_rcp(fmaxf(fast_sqrt(ss2), 1e-12f));

        float l1 = 0.0f;
        #pragma unroll
        for (int i = 0; i < 8; ++i) {
            d1[i] *= inv2;
            l1 += fabsf(d1[i]);
        }
        l1 = reduce16(l1);
        const float invl1 = fast_rcp(fmaxf(l1, 1e-12f));

        // descriptor index = bin*16 + (cy*4+cx); acc[v] is bin v, col = lane%16.
        // Streaming output: nontemporal stores keep near caches for the input.
        float* __restrict__ o = out + ((size_t)pbase + (hi ? 1 : 0)) * 128;
        #pragma unroll
        for (int v = 0; v < 8; ++v)
            __builtin_nontemporal_store(fast_sqrt(d1[v] * invl1 + EPSV),
                                        &o[v * 16 + half16]);
    }
}

extern "C" void kernel_launch(void* const* d_in, const int* in_sizes, int n_in,
                              void* d_out, int out_size, void* d_ws, size_t ws_size,
                              hipStream_t stream) {
    (void)n_in; (void)d_ws; (void)ws_size; (void)out_size;
    const float* in = (const float*)d_in[0];
    float* out = (float*)d_out;
    const int nPatches = in_sizes[0] / (PATCH * PATCH);
    if (nPatches < 2) return;
    const int totalPairs = nPatches >> 1;
    int blocks = (totalPairs + 7) / 8;     // 8 pairs per block-step
    if (blocks > 256) blocks = 256;        // grid-stride: ~4 pairs per wave at B=16384
    SIFTDescriptor_71983651881333_kernel<<<blocks, 256, 0, stream>>>(in, out, nPatches);
}
